// AdaptiveAttentionLayer_45183055954172
// MI455X (gfx1250) — matrix operand order repaired and checked by reading, not compiled, not run on hardware
//
#include <hip/hip_runtime.h>
#include <hip/hip_bf16.h>

typedef __attribute__((ext_vector_type(16))) __bf16 v16bf;
typedef __attribute__((ext_vector_type(8)))  __bf16 v8bf;
typedef __attribute__((ext_vector_type(8)))  float  v8f;

// ---------------------------------------------------------------------------
// WMMA helpers (gfx1250, wave32).  D = A(16x32 bf16) * B(32x16 bf16) + C(f32)
// ---------------------------------------------------------------------------
__device__ __forceinline__ v8f wmma_bf16(v16bf a, v16bf b, v8f c) {
  return __builtin_amdgcn_wmma_f32_16x16x32_bf16(false, a, false, b, (short)0, c,
                                                 false, false);
}

// A-matrix 16x32 bf16 fragment (ISA 7.12.2): lanes 0-15: M=lane, K=0..7 & 16..23;
// lanes 16-31: M=lane-16, K=8..15 & 24..31.  src row-major, leading dim `ld`.
__device__ __forceinline__ v16bf load_frag_a(const __bf16* base, int ld) {
  int lane = threadIdx.x & 31;
  const __bf16* p = base + (size_t)(lane & 15) * ld + ((lane >> 4) << 3);
  v8bf lo = *(const v8bf*)p;
  v8bf hi = *(const v8bf*)(p + 16);
  v16bf r;
#pragma unroll
  for (int i = 0; i < 8; ++i) { r[i] = lo[i]; r[i + 8] = hi[i]; }
  return r;
}

// B-matrix 32x16 bf16 fragment: lane n (0-15) holds column n, K=0..7 & 16..23;
// lane n+16 holds column n, K=8..15 & 24..31.  Element (k, col) at base[col*ld + k].
__device__ __forceinline__ v16bf load_frag_b(const __bf16* base, int ld) {
  int lane = threadIdx.x & 31;
  const __bf16* p = base + (size_t)(lane & 15) * ld + ((lane >> 4) << 3);
  v8bf lo = *(const v8bf*)p;
  v8bf hi = *(const v8bf*)(p + 16);
  v16bf r;
#pragma unroll
  for (int i = 0; i < 8; ++i) { r[i] = lo[i]; r[i + 8] = hi[i]; }
  return r;
}

#define BB 4
#define CC 256
#define NN 4096

// ---------------------------------------------------------------------------
// Kernel 1: instance norm over spatial axes per (b,c).  grid=(B*C, 2)
// ---------------------------------------------------------------------------
__global__ void instnorm_kernel(const float* __restrict__ content,
                                const float* __restrict__ style,
                                float* __restrict__ normCf,
                                __bf16* __restrict__ normCb,
                                __bf16* __restrict__ normSb,
                                __bf16* __restrict__ styleb) {
  int which = blockIdx.y;
  const float* src = which ? style : content;
  int bc = blockIdx.x;
  int b = bc >> 8, c = bc & 255;
  const float* base = src + (size_t)b * NN * CC + c;
  int tid = threadIdx.x;

  float s = 0.f, sq = 0.f;
  for (int i = tid; i < NN; i += 256) {
    float x = base[(size_t)i * CC];
    s += x; sq += x * x;
  }
  __shared__ float rs[256], rq[256];
  rs[tid] = s; rq[tid] = sq;
  __syncthreads();
  for (int o = 128; o > 0; o >>= 1) {
    if (tid < o) { rs[tid] += rs[tid + o]; rq[tid] += rq[tid + o]; }
    __syncthreads();
  }
  float mean = rs[0] * (1.0f / NN);
  float var = rq[0] * (1.0f / NN) - mean * mean;
  float rinv = rsqrtf(fmaxf(var, 0.f) + 1e-5f);

  for (int i = tid; i < NN; i += 256) {
    float x = base[(size_t)i * CC];
    float y = (x - mean) * rinv;
    size_t idx = ((size_t)b * NN + i) * CC + c;
    if (which == 0) { normCf[idx] = y; normCb[idx] = (__bf16)y; }
    else            { normSb[idx] = (__bf16)y; styleb[idx] = (__bf16)x; }
  }
}

// ---------------------------------------------------------------------------
// Kernel 2: W[c][d] f32 -> Wt[d][c] bf16.  grid=(256, 3)
// ---------------------------------------------------------------------------
__global__ void convw_kernel(const float* __restrict__ Wq,
                             const float* __restrict__ Wk,
                             const float* __restrict__ Wv,
                             __bf16* __restrict__ Wtq,
                             __bf16* __restrict__ Wtk,
                             __bf16* __restrict__ Wtv) {
  int which = blockIdx.y;
  const float* Win = (which == 0) ? Wq : (which == 1) ? Wk : Wv;
  __bf16* Wt = (which == 0) ? Wtq : (which == 1) ? Wtk : Wtv;
  int idx = blockIdx.x * 256 + threadIdx.x;   // idx = d*256 + c
  int d = idx >> 8, c = idx & 255;
  Wt[idx] = (__bf16)Win[c * 256 + d];
}

// ---------------------------------------------------------------------------
// Kernel 3: projection GEMM  Y = X[16384x256] * W + b.
//   mode 0: row-l2-normalize, write bf16 row-major (Q / K)
//   mode 1: write Vt / V2t channel-major bf16 ([b][chan][n])
// ---------------------------------------------------------------------------
__global__ void proj_kernel(const __bf16* __restrict__ X,
                            const __bf16* __restrict__ Wt,
                            const float* __restrict__ bias,
                            __bf16* __restrict__ outA,
                            __bf16* __restrict__ outB, int mode) {
  int tid = threadIdx.x, lane = tid & 31, wave = tid >> 5;
  int n16 = lane & 15, hi = lane >> 4;
  size_t row0 = (size_t)blockIdx.x * 16;
  int chan0 = wave * 32;

  v8f c0 = {}, c1 = {};
  const __bf16* Xb = X + row0 * CC;
  const __bf16* W0 = Wt + (size_t)chan0 * CC;
  const __bf16* W1 = Wt + (size_t)(chan0 + 16) * CC;
#pragma unroll
  for (int kk = 0; kk < 8; ++kk) {
    v16bf a  = load_frag_a(Xb + kk * 32, CC);
    v16bf b0 = load_frag_b(W0 + kk * 32, CC);
    v16bf b1 = load_frag_b(W1 + kk * 32, CC);
    c0 = wmma_bf16(a, b0, c0);
    c1 = wmma_bf16(a, b1, c1);
  }
  float bi0 = bias[chan0 + n16];
  float bi1 = bias[chan0 + 16 + n16];

  if (mode == 0) {
    __shared__ float ss[16];
    if (tid < 16) ss[tid] = 0.f;
    __syncthreads();
#pragma unroll
    for (int r = 0; r < 8; ++r) {
      float v0 = c0[r] + bi0, v1 = c1[r] + bi1;
      atomicAdd(&ss[r + 8 * hi], v0 * v0 + v1 * v1);
    }
    __syncthreads();
#pragma unroll
    for (int r = 0; r < 8; ++r) {
      int row = r + 8 * hi;
      float inv = rsqrtf(ss[row] + 1e-12f);
      size_t base = (row0 + row) * CC;
      outA[base + chan0 + n16]      = (__bf16)((c0[r] + bi0) * inv);
      outA[base + chan0 + 16 + n16] = (__bf16)((c1[r] + bi1) * inv);
    }
  } else {
    size_t bb = row0 >> 12;        // N = 4096
    size_t nloc = row0 & 4095;
#pragma unroll
    for (int r = 0; r < 8; ++r) {
      int row = r + 8 * hi;
      float v0 = c0[r] + bi0, v1 = c1[r] + bi1;
      size_t i0 = ((bb * CC) + chan0 + n16) * (size_t)NN + nloc + row;
      size_t i1 = ((bb * CC) + chan0 + 16 + n16) * (size_t)NN + nloc + row;
      outA[i0] = (__bf16)v0;        outA[i1] = (__bf16)v1;
      outB[i0] = (__bf16)(v0 * v0); outB[i1] = (__bf16)(v1 * v1);
    }
  }
}

// ---------------------------------------------------------------------------
// Kernel 4: flash attention, 32 query rows per block (2 row-sets of 16),
// dual accumulators (M = A*V, S2 = A*V^2), fused finalize.
// Q staged in LDS; every K/V/V^2 B-fragment feeds 2 WMMAs (1:1 b128:wmma).
// grid = (N/32, B), 8 waves; wave w owns channels [32w,32w+32) and the
// logits for its own 32 keys of each 256-key chunk.
// ---------------------------------------------------------------------------
__global__ void attn_kernel(const __bf16* __restrict__ Qb,
                            const __bf16* __restrict__ Kb,
                            const __bf16* __restrict__ Vt,
                            const __bf16* __restrict__ V2t,
                            const float* __restrict__ normCf,
                            float* __restrict__ out) {
  int tid = threadIdx.x, lane = tid & 31, wave = tid >> 5;
  int n16 = lane & 15, hi = lane >> 4;
  int row0 = blockIdx.x * 32;
  int bb = blockIdx.y;

  __shared__ float row_m[32], row_l[32], row_alpha[32];
  __shared__ float sm_max[8][32], sm_sum[8][32];
  __shared__ __align__(16) __bf16 Ql[32 * 256];
  __shared__ __align__(16) __bf16 Pl[32 * 256];

  if (tid < 32) { row_m[tid] = -1e30f; row_l[tid] = 0.f; }

  // stage the 32x256 Q row-block into LDS (contiguous 16 KB copy)
  {
    const v8bf* src = (const v8bf*)(Qb + ((size_t)bb * NN + row0) * CC);
    v8bf* dst = (v8bf*)&Ql[0];
    for (int i = tid; i < 32 * 256 / 8; i += 256) dst[i] = src[i];
  }

  v8f accM[2][2] = {};   // [rowset][chan tile]
  v8f accS[2][2] = {};
  const __bf16* Vbase  = Vt  + ((size_t)bb * CC + wave * 32) * NN;
  const __bf16* V2base = V2t + ((size_t)bb * CC + wave * 32) * NN;
  __syncthreads();

  for (int kc = 0; kc < NN; kc += 256) {
    // ---- logits: 2 row-sets x 2 key-tiles, shared A/B fragments ----
    const __bf16* Kbase = Kb + ((size_t)bb * NN + kc + wave * 32) * CC;
    v8f c[2][2] = {};
#pragma unroll
    for (int kk = 0; kk < 8; ++kk) {
      v16bf a0 = load_frag_a(&Ql[0] + kk * 32, 256);
      v16bf a1 = load_frag_a(&Ql[0] + 16 * 256 + kk * 32, 256);
      v16bf b0 = load_frag_b(Kbase + kk * 32, CC);
      v16bf b1 = load_frag_b(Kbase + (size_t)16 * CC + kk * 32, CC);
      c[0][0] = wmma_bf16(a0, b0, c[0][0]);
      c[0][1] = wmma_bf16(a0, b1, c[0][1]);
      c[1][0] = wmma_bf16(a1, b0, c[1][0]);
      c[1][1] = wmma_bf16(a1, b1, c[1][1]);
    }
    // ---- per-row max over this wave's 32 keys ----
    float tm[2][8];
#pragma unroll
    for (int rs = 0; rs < 2; ++rs)
#pragma unroll
      for (int r = 0; r < 8; ++r) tm[rs][r] = fmaxf(c[rs][0][r], c[rs][1][r]);
#pragma unroll
    for (int m = 1; m < 16; m <<= 1) {
#pragma unroll
      for (int rs = 0; rs < 2; ++rs)
#pragma unroll
        for (int r = 0; r < 8; ++r)
          tm[rs][r] = fmaxf(tm[rs][r], __shfl_xor(tm[rs][r], m));
    }
    if (n16 == 0) {
#pragma unroll
      for (int rs = 0; rs < 2; ++rs)
#pragma unroll
        for (int r = 0; r < 8; ++r)
          sm_max[wave][16 * rs + r + 8 * hi] = tm[rs][r];
    }
    __syncthreads();
    if (tid < 32) {
      float cm = sm_max[0][tid];
#pragma unroll
      for (int w = 1; w < 8; ++w) cm = fmaxf(cm, sm_max[w][tid]);
      float mo = row_m[tid];
      float mn = fmaxf(mo, cm);
      row_alpha[tid] = __expf(mo - mn);
      row_m[tid] = mn;
    }
    __syncthreads();
    // ---- exp, stage P in LDS (bf16), partial row sums ----
    float ps[2][8];
#pragma unroll
    for (int rs = 0; rs < 2; ++rs) {
#pragma unroll
      for (int r = 0; r < 8; ++r) {
        int row = 16 * rs + r + 8 * hi;
        float mn = row_m[row];
        float p0 = __expf(c[rs][0][r] - mn);
        float p1 = __expf(c[rs][1][r] - mn);
        ps[rs][r] = p0 + p1;
        Pl[row * 256 + wave * 32 + n16]      = (__bf16)p0;
        Pl[row * 256 + wave * 32 + 16 + n16] = (__bf16)p1;
      }
    }
#pragma unroll
    for (int m = 1; m < 16; m <<= 1) {
#pragma unroll
      for (int rs = 0; rs < 2; ++rs)
#pragma unroll
        for (int r = 0; r < 8; ++r) ps[rs][r] += __shfl_xor(ps[rs][r], m);
    }
    if (n16 == 0) {
#pragma unroll
      for (int rs = 0; rs < 2; ++rs)
#pragma unroll
        for (int r = 0; r < 8; ++r)
          sm_sum[wave][16 * rs + r + 8 * hi] = ps[rs][r];
    }
    __syncthreads();                      // Pl complete, sums posted
    if (tid < 32) {
      float s = 0.f;
#pragma unroll
      for (int w = 0; w < 8; ++w) s += sm_sum[w][tid];
      row_l[tid] = row_l[tid] * row_alpha[tid] + s;
    }
    // ---- rescale accumulators, then P @ V and P @ V^2 ----
#pragma unroll
    for (int rs = 0; rs < 2; ++rs) {
#pragma unroll
      for (int r = 0; r < 8; ++r) {
        float a = row_alpha[16 * rs + r + 8 * hi];
        accM[rs][0][r] *= a; accM[rs][1][r] *= a;
        accS[rs][0][r] *= a; accS[rs][1][r] *= a;
      }
    }
#pragma unroll
    for (int kk = 0; kk < 8; ++kk) {
      v16bf pa0 = load_frag_a(&Pl[0] + kk * 32, 256);
      v16bf pa1 = load_frag_a(&Pl[0] + 16 * 256 + kk * 32, 256);
      const __bf16* vb = Vbase + kc + kk * 32;
      v16bf bv0 = load_frag_b(vb, NN);
      v16bf bv1 = load_frag_b(vb + (size_t)16 * NN, NN);
      accM[0][0] = wmma_bf16(pa0, bv0, accM[0][0]);
      accM[0][1] = wmma_bf16(pa0, bv1, accM[0][1]);
      accM[1][0] = wmma_bf16(pa1, bv0, accM[1][0]);
      accM[1][1] = wmma_bf16(pa1, bv1, accM[1][1]);
      const __bf16* v2b = V2base + kc + kk * 32;
      v16bf d0 = load_frag_b(v2b, NN);
      v16bf d1 = load_frag_b(v2b + (size_t)16 * NN, NN);
      accS[0][0] = wmma_bf16(pa0, d0, accS[0][0]);
      accS[0][1] = wmma_bf16(pa0, d1, accS[0][1]);
      accS[1][0] = wmma_bf16(pa1, d0, accS[1][0]);
      accS[1][1] = wmma_bf16(pa1, d1, accS[1][1]);
    }
  }

  __syncthreads();                        // row_l final
#pragma unroll
  for (int rs = 0; rs < 2; ++rs) {
#pragma unroll
    for (int r = 0; r < 8; ++r) {
      int row = 16 * rs + r + 8 * hi;
      float linv = 1.0f / row_l[row];
#pragma unroll
      for (int t = 0; t < 2; ++t) {
        float Mv = accM[rs][t][r] * linv;
        float s2 = accS[rs][t][r] * linv - Mv * Mv;
        float S = (s2 > 0.f) ? sqrtf(s2) : 0.f;
        size_t oi = ((size_t)bb * NN + row0 + row) * CC + wave * 32 + 16 * t + n16;
        out[oi] = S * normCf[oi] + Mv;
      }
    }
  }
}

// ---------------------------------------------------------------------------
extern "C" void kernel_launch(void* const* d_in, const int* in_sizes, int n_in,
                              void* d_out, int out_size, void* d_ws, size_t ws_size,
                              hipStream_t stream) {
  const float* content = (const float*)d_in[0];
  const float* style   = (const float*)d_in[1];
  const float* Wq = (const float*)d_in[2];
  const float* bq = (const float*)d_in[3];
  const float* Wk = (const float*)d_in[4];
  const float* bk = (const float*)d_in[5];
  const float* Wv = (const float*)d_in[6];
  const float* bv = (const float*)d_in[7];
  float* out = (float*)d_out;

  const size_t BNC = (size_t)BB * NN * CC;   // 4,194,304 elements
  char* w = (char*)d_ws;
  float*  normCf = (float*)w;   w += BNC * sizeof(float);
  __bf16* normCb = (__bf16*)w;  w += BNC * 2;
  __bf16* normSb = (__bf16*)w;  w += BNC * 2;
  __bf16* styleb = (__bf16*)w;  w += BNC * 2;
  __bf16* Qb     = (__bf16*)w;  w += BNC * 2;
  __bf16* Kbuf   = (__bf16*)w;  w += BNC * 2;
  __bf16* Vt     = (__bf16*)w;  w += BNC * 2;
  __bf16* V2t    = (__bf16*)w;  w += BNC * 2;
  __bf16* Wtq    = (__bf16*)w;  w += (size_t)CC * CC * 2;
  __bf16* Wtk    = (__bf16*)w;  w += (size_t)CC * CC * 2;
  __bf16* Wtv    = (__bf16*)w;  w += (size_t)CC * CC * 2;

  instnorm_kernel<<<dim3(BB * CC, 2), 256, 0, stream>>>(
      content, style, normCf, normCb, normSb, styleb);
  convw_kernel<<<dim3(256, 3), 256, 0, stream>>>(Wq, Wk, Wv, Wtq, Wtk, Wtv);
  proj_kernel<<<dim3(BB * NN / 16), 256, 0, stream>>>(normCb, Wtq, bq, Qb, nullptr, 0);
  proj_kernel<<<dim3(BB * NN / 16), 256, 0, stream>>>(normSb, Wtk, bk, Kbuf, nullptr, 0);
  proj_kernel<<<dim3(BB * NN / 16), 256, 0, stream>>>(styleb, Wtv, bv, Vt, V2t, 1);
  attn_kernel<<<dim3(NN / 32, BB), 256, 0, stream>>>(Qb, Kbuf, Vt, V2t, normCf, out);
}
